// MoEDeepSeekV3_5282809775022
// MI455X (gfx1250) — matrix operand order repaired and checked by reading, not compiled, not run on hardware
//
#include <hip/hip_runtime.h>
#include <cstdint>
#include <cstddef>

// ---------------- problem constants (match reference) ----------------
#define TTOK    8192
#define DIMC    2048
#define NEXP    32
#define TOPKC   6
#define NGRP    8
#define TOPG    4
#define INTERC  1408
#define SINTERC 2816
#define RSCALE  2.5f

// ---------------- tiling ----------------
#define BLKT    512                 // 16 waves per block
#define CHUNK2  256                 // inter-dim chunk (16 waves x 16)
#define NTIL2   8                   // GEMM2 output tiles per wave (128 cols)
#define XS      2056                // bf16 row stride of X tile (pad, mult of 8)
#define HSTR2   264                 // bf16 row stride of H chunk (pad, mult of 8)
#define XBYTES  (16 * XS * 2)       // 65792
#define HBYTES  (16 * HSTR2 * 2)    // 8448
#define SMEM_BYTES (XBYTES + HBYTES + 64 + 64)   // 74368 (<320KB WGP LDS)

typedef __attribute__((ext_vector_type(16))) __bf16       v16bf;
typedef __attribute__((ext_vector_type(8)))  float        v8f;
typedef __attribute__((ext_vector_type(4)))  unsigned int u32x4;
typedef __attribute__((ext_vector_type(4)))  float        f32x4;

union Frag {
    v16bf  v;
    u32x4  q[2];
    __bf16 h[16];
};

static __device__ inline v8f vzero8() {
    v8f r;
#pragma unroll
    for (int i = 0; i < 8; ++i) r[i] = 0.f;
    return r;
}

// A-matrix fragment (16x32 bf16) from an LDS tile, in contiguous k order (same order as B):
//   lanes 0-15: row M=lane, halves 0..15 = K kb+0..15
//   lanes 16-31: row M=lane-16, halves 0..15 = K kb+16..31
static __device__ inline v16bf load_a_frag_lds(const __bf16* lds, int rowStride,
                                               int kbase, int lane) {
    int row = lane & 15;
    int hi  = lane >> 4;
    const __bf16* p = lds + row * rowStride + kbase + hi * 16;
    Frag f;
    f.q[0] = *(const u32x4*)(p);
    f.q[1] = *(const u32x4*)(p + 8);
    return f.v;
}

// Raw (fp32) B-matrix fragment load: 64B per lane from row-major W[nRows][ldw],
// B[k][n] = W[nbase+n][kbase+k]; lanes 0-15 hold K kb+0..15 (col = lane&15),
// lanes 16-31 hold K kb+16..31.
static __device__ inline void load_b_raw(f32x4 r[4], const float* __restrict__ W,
                                         int ldw, int nbase, int kbase, int lane) {
    int n  = nbase + (lane & 15);
    int hi = lane >> 4;
    const f32x4* p4 = (const f32x4*)(W + (size_t)n * (size_t)ldw + (size_t)(kbase + hi * 16));
#pragma unroll
    for (int i = 0; i < 4; ++i) r[i] = p4[i];
}

static __device__ inline v16bf cvt_b(const f32x4 r[4]) {
    Frag f;
#pragma unroll
    for (int i = 0; i < 4; ++i) {
        f.h[i * 4 + 0] = (__bf16)r[i][0];
        f.h[i * 4 + 1] = (__bf16)r[i][1];
        f.h[i * 4 + 2] = (__bf16)r[i][2];
        f.h[i * 4 + 3] = (__bf16)r[i][3];
    }
    return f.v;
}

static __device__ inline v16bf load_b_frag_f32(const float* __restrict__ W, int ldw,
                                               int nbase, int kbase, int lane) {
    f32x4 r[4];
    load_b_raw(r, W, ldw, nbase, kbase, lane);
    return cvt_b(r);
}

// ---------------------------------------------------------------------
// Kernel 1: sigmoid gate + group-limited top-k routing. One wave per token.
// ---------------------------------------------------------------------
__global__ __launch_bounds__(256)
void gate_topk(const float* __restrict__ x, const float* __restrict__ gw,
               int* __restrict__ topkIdx, float* __restrict__ topkW) {
    __shared__ float sc[8][NEXP];
    __shared__ float msc[8][NEXP];
    int wave = threadIdx.x >> 5;
    int lane = threadIdx.x & 31;
    int t = blockIdx.x * 8 + wave;

    const float* xr = x + (size_t)t * DIMC;
    for (int e = 0; e < NEXP; ++e) {
        const float* wr = gw + (size_t)e * DIMC;
        float acc = 0.f;
        for (int d = lane; d < DIMC; d += 32) acc += xr[d] * wr[d];
#pragma unroll
        for (int off = 16; off > 0; off >>= 1) acc += __shfl_down(acc, off);
        if (lane == 0) sc[wave][e] = 1.f / (1.f + __expf(-acc));
    }

    if (lane == 0) {
        float gm[NGRP];
#pragma unroll
        for (int g = 0; g < NGRP; ++g) {
            float m = sc[wave][g * 4];
#pragma unroll
            for (int j = 1; j < 4; ++j) m = fmaxf(m, sc[wave][g * 4 + j]);
            gm[g] = m;
        }
        bool gsel[NGRP] = {false, false, false, false, false, false, false, false};
        for (int r = 0; r < TOPG; ++r) {
            int best = 0; float bv = -1e30f;
            for (int g = 0; g < NGRP; ++g)
                if (!gsel[g] && gm[g] > bv) { bv = gm[g]; best = g; }
            gsel[best] = true;
        }
        for (int e = 0; e < NEXP; ++e)
            msc[wave][e] = gsel[e >> 2] ? sc[wave][e] : -1.f;
        int   idx[TOPKC];
        float wk[TOPKC];
        float wsum = 0.f;
        for (int k = 0; k < TOPKC; ++k) {
            int best = 0; float bv = -1e30f;
            for (int e = 0; e < NEXP; ++e) {
                float v = msc[wave][e];
                if (v > bv) { bv = v; best = e; }
            }
            idx[k] = best; wk[k] = sc[wave][best]; wsum += wk[k];
            msc[wave][best] = -2e30f;
        }
        float scale = RSCALE / wsum;
        for (int k = 0; k < TOPKC; ++k) {
            topkIdx[(size_t)t * TOPKC + k] = idx[k];
            topkW[(size_t)t * TOPKC + k]   = wk[k] * scale;
        }
    }
}

// ---------------------------------------------------------------------
// Kernel 2: deterministic per-expert token compaction (one block / expert).
// ---------------------------------------------------------------------
__global__ __launch_bounds__(256)
void build_buckets(const int* __restrict__ topkIdx, const float* __restrict__ topkW,
                   int* __restrict__ cnt, int* __restrict__ bucketTok,
                   float* __restrict__ bucketW) {
    __shared__ int cntArr[256];
    __shared__ int base[257];
    int e = blockIdx.x;
    int tid = threadIdx.x;
    const int TPT = TTOK / 256;
    int t0 = tid * TPT;

    int m = 0;
    for (int t = t0; t < t0 + TPT; ++t)
#pragma unroll
        for (int k = 0; k < TOPKC; ++k)
            if (topkIdx[(size_t)t * TOPKC + k] == e) ++m;
    cntArr[tid] = m;
    __syncthreads();
    if (tid == 0) {
        int acc = 0;
        for (int i = 0; i < 256; ++i) { base[i] = acc; acc += cntArr[i]; }
        base[256] = acc;
        cnt[e] = acc;
    }
    __syncthreads();
    int pos = base[tid];
    for (int t = t0; t < t0 + TPT; ++t)
#pragma unroll
        for (int k = 0; k < TOPKC; ++k)
            if (topkIdx[(size_t)t * TOPKC + k] == e) {
                bucketTok[(size_t)e * TTOK + pos] = t;
                bucketW[(size_t)e * TTOK + pos]   = topkW[(size_t)t * TOPKC + k];
                ++pos;
            }
}

// ---------------------------------------------------------------------
// Kernel 3: fused SwiGLU FFN tile via bf16 WMMA, fp32 accumulate.
//   block = 512 threads (16 waves), M = 16 tokens.
//   16 waves -> GEMM2 accumulators are only 64 VGPRs/wave (8 tiles x v8f),
//   keeping total pressure < 256 VGPRs and occupancy at ~4 waves/SIMD,
//   which is what hides the streamed (zero-reuse) weight load latency.
// ---------------------------------------------------------------------
__global__ __launch_bounds__(BLKT)
void ffn_swiglu_wmma(const float* __restrict__ x,
                     const float* __restrict__ W1,   // [inter, DIM] (per expert)
                     const float* __restrict__ W3,   // [inter, DIM]
                     const float* __restrict__ W2,   // [DIM, inter]
                     float* __restrict__ out,
                     const int* __restrict__ cnt,
                     const int* __restrict__ bucketTok,
                     const float* __restrict__ bucketW,
                     int inter, int accumulate) {
    extern __shared__ char smem[];
    __bf16* Xlds = (__bf16*)smem;
    __bf16* Hlds = (__bf16*)(smem + XBYTES);
    float*  cwT  = (float*)(smem + XBYTES + HBYTES);
    int*    tokT = (int*)(smem + XBYTES + HBYTES + 64);

    int tid  = threadIdx.x;
    int wave = tid >> 5;       // 0..15
    int lane = tid & 31;
    int hi   = lane >> 4;
    int ln   = lane & 15;
    int tileBase = blockIdx.x * 16;

    int e = 0, count = TTOK, gather = 0;
    const float *W1e = W1, *W3e = W3, *W2e = W2;
    if (cnt) {
        e = blockIdx.y;
        count = cnt[e];
        if (tileBase >= count) return;      // empty tile for this expert
        W1e += (size_t)e * INTERC * DIMC;
        W3e += (size_t)e * INTERC * DIMC;
        W2e += (size_t)e * DIMC * INTERC;
        gather = 1;
    }

    // stage token ids + combine weights for this 16-row tile
    if (tid < 16) {
        int row = tid, tok; float w;
        if (gather) {
            int gi = tileBase + row;
            if (gi < count) {
                tok = bucketTok[(size_t)e * TTOK + gi];
                w   = bucketW[(size_t)e * TTOK + gi];
            } else { tok = 0; w = 0.f; }     // padded row: weight 0
        } else { tok = tileBase + row; w = 1.f; }
        tokT[row] = tok;
        cwT[row]  = w;
    }
    __syncthreads();

    // stage X tile (16 x 2048) fp32 -> bf16 in LDS (coalesced)
#pragma unroll 4
    for (int i = tid; i < 16 * DIMC; i += BLKT) {
        int row = i >> 11;
        int col = i & (DIMC - 1);
        Xlds[row * XS + col] = (__bf16)x[(size_t)tokT[row] * DIMC + col];
    }
    __syncthreads();

    // fp32 output accumulators: this wave owns N slice [wave*128, wave*128+128)
    v8f yacc[NTIL2];
#pragma unroll
    for (int t = 0; t < NTIL2; ++t) yacc[t] = vzero8();

    for (int chunkBase = 0; chunkBase < inter; chunkBase += CHUNK2) {
        int chunkK = inter - chunkBase;
        if (chunkK > CHUNK2) chunkK = CHUNK2;   // 256, or 128 tail (routed)

        // ---- GEMM1: h = silu(X W1^T) * (X W3^T); wave owns 16-wide slice
        int interN = chunkBase + wave * 16;
        if (wave * 16 < chunkK) {
            v8f p1 = vzero8();
            v8f p3 = vzero8();
#pragma unroll 2
            for (int ks = 0; ks < DIMC / 32; ++ks) {
                int kb = ks * 32;
                f32x4 r1[4], r3[4];
                load_b_raw(r1, W1e, DIMC, interN, kb, lane);   // one 8-load clause
                load_b_raw(r3, W3e, DIMC, interN, kb, lane);
                v16bf a  = load_a_frag_lds(Xlds, XS, kb, lane);
                v16bf b1 = cvt_b(r1);
                p1 = __builtin_amdgcn_wmma_f32_16x16x32_bf16(false, a, false, b1,
                                                             (short)0, p1, false, false);
                v16bf b3 = cvt_b(r3);
                p3 = __builtin_amdgcn_wmma_f32_16x16x32_bf16(false, a, false, b3,
                                                             (short)0, p3, false, false);
            }
            // fuse SiLU * gate, write H slice (16 rows x 16 cols bf16) to LDS
#pragma unroll
            for (int j = 0; j < 8; ++j) {
                int row = j + hi * 8;                // C/D layout: M = j + 8*hi
                float g = p1[j];
                float hval = (g / (1.f + __expf(-g))) * p3[j];
                Hlds[row * HSTR2 + wave * 16 + ln] = (__bf16)hval;
            }
        }
        __syncthreads();

        // ---- GEMM2: y += H W2c^T over this chunk (K = chunkK)
        int kSteps = chunkK >> 5;                    // 8 (full) or 4 (tail)
        for (int ks = 0; ks < kSteps; ++ks) {
            int kb = ks * 32;
            v16bf ah = load_a_frag_lds(Hlds, HSTR2, kb, lane);
#pragma unroll
            for (int t = 0; t < NTIL2; ++t) {
                v16bf b2 = load_b_frag_f32(W2e, inter, wave * 128 + t * 16,
                                           chunkBase + kb, lane);
                yacc[t] = __builtin_amdgcn_wmma_f32_16x16x32_bf16(false, ah, false, b2,
                                                                  (short)0, yacc[t],
                                                                  false, false);
            }
        }
        __syncthreads();   // protect H before next chunk overwrites it
    }

    // writeout: scale by combine weight, scatter to token rows
#pragma unroll
    for (int t = 0; t < NTIL2; ++t) {
        int n = wave * 128 + t * 16 + ln;
#pragma unroll
        for (int j = 0; j < 8; ++j) {
            int row = j + hi * 8;
            float v = yacc[t][j] * cwT[row];
            size_t addr = (size_t)tokT[row] * DIMC + (size_t)n;
            if (accumulate) atomicAdd(out + addr, v);
            else            out[addr] = v;
        }
    }
}

// ---------------------------------------------------------------------
extern "C" void kernel_launch(void* const* d_in, const int* in_sizes, int n_in,
                              void* d_out, int out_size, void* d_ws, size_t ws_size,
                              hipStream_t stream) {
    (void)in_sizes; (void)n_in; (void)out_size; (void)ws_size;
    const float* x   = (const float*)d_in[0];
    const float* gw  = (const float*)d_in[1];
    const float* w1  = (const float*)d_in[2];
    const float* w2  = (const float*)d_in[3];
    const float* w3  = (const float*)d_in[4];
    const float* sw1 = (const float*)d_in[5];
    const float* sw2 = (const float*)d_in[6];
    const float* sw3 = (const float*)d_in[7];
    float* out = (float*)d_out;

    // workspace layout (~2.5 MB)
    char* ws = (char*)d_ws;
    int*   cnt       = (int*)ws;                                     // 32 ints
    int*   topkIdx   = (int*)(ws + 256);                             // T*6 ints
    float* topkW     = (float*)(ws + 256 + (size_t)TTOK * TOPKC * 4);
    int*   bucketTok = (int*)(ws + 256 + 2 * (size_t)TTOK * TOPKC * 4);
    float* bucketW   = (float*)(ws + 256 + 2 * (size_t)TTOK * TOPKC * 4 +
                                (size_t)NEXP * TTOK * 4);

    // allow >64KB dynamic LDS (WGP has 320KB)
    (void)hipFuncSetAttribute((const void*)ffn_swiglu_wmma,
                              hipFuncAttributeMaxDynamicSharedMemorySize,
                              SMEM_BYTES);

    // 1) gate + routing (one wave per token)
    gate_topk<<<TTOK / 8, 256, 0, stream>>>(x, gw, topkIdx, topkW);
    // 2) deterministic per-expert compaction
    build_buckets<<<NEXP, 256, 0, stream>>>(topkIdx, topkW, cnt, bucketTok, bucketW);
    // 3) shared experts: plain store (writes every out element exactly once)
    ffn_swiglu_wmma<<<dim3(TTOK / 16, 1), BLKT, SMEM_BYTES, stream>>>(
        x, sw1, sw3, sw2, out, nullptr, nullptr, nullptr, SINTERC, 0);
    // 4) routed experts: atomic accumulate on top (worst-case grid, early exit)
    ffn_swiglu_wmma<<<dim3(TTOK / 16, NEXP), BLKT, SMEM_BYTES, stream>>>(
        x, w1, w3, w2, out, cnt, bucketTok, bucketW, INTERC, 1);
}
